// EmbeddingsModel_76965813944901
// MI455X (gfx1250) — matrix operand order repaired and checked by reading, not compiled, not run on hardware
//
#include <hip/hip_runtime.h>
#include <hip/hip_bf16.h>

typedef __attribute__((ext_vector_type(16))) __bf16 v16bf;
typedef __attribute__((ext_vector_type(8)))  __bf16 v8bf;
typedef __attribute__((ext_vector_type(8)))  float  v8f;

#define NUM_NODES 100000
#define EMB       128
#define BATCH     1024

// Split f32 into bf16 hi + bf16 lo residual (combined error ~2^-18 relative).
__device__ __forceinline__ void split_bf16(float x, __bf16& hi, __bf16& lo)
{
    hi = (__bf16)x;
    lo = (__bf16)(x - (float)hi);
}

// ---------------------------------------------------------------------------
// Stage 1: combined[b][e] = embeddings[subj[b]][e] * relations[rel[b]][e],
// pre-split into bf16 hi/lo planes so the GEMM inner loop does ZERO
// conversion VALU for A. 2 x 256 KB in workspace, L2-resident.
// ---------------------------------------------------------------------------
__global__ void __launch_bounds__(256)
distmult_combine_split(const int* __restrict__ subj,
                       const int* __restrict__ rel,
                       const float* __restrict__ emb,
                       const float* __restrict__ rels,
                       __bf16* __restrict__ Ahi,
                       __bf16* __restrict__ Alo)
{
    int idx = blockIdx.x * 256 + threadIdx.x;   // 0 .. BATCH*EMB-1
    int b = idx >> 7;                           // / EMB
    int e = idx & (EMB - 1);
    int s = subj[b];
    int r = rel[b];
    float x = emb[(size_t)s * EMB + e] * rels[(size_t)r * EMB + e];
    __bf16 h, l;
    split_bf16(x, h, l);
    Ahi[idx] = h;
    Alo[idx] = l;
}

// ---------------------------------------------------------------------------
// Stage 2: scores[M=1024][N=100000] = A[M][K=128] . embeddings[N][K]^T
// Grid: 6250 blocks (one 16-wide N strip each), 256 threads = 8 waves.
// Wave w computes M rows [w*128, w*128+128) = 8 tiles of 16x16.
// Per K-chunk of 32: 3x v_wmma_f32_16x16x32_bf16 (hi*hi + hi*lo + lo*hi).
// B is converted to hi/lo bf16 ONCE per wave (outside the tile loop);
// A hi/lo planes are loaded pre-split as raw b128s.
// ---------------------------------------------------------------------------
__global__ void __launch_bounds__(256)
distmult_gemm(const float* __restrict__ emb,     // [NUM_NODES][EMB] f32
              const __bf16* __restrict__ Ahi,    // [BATCH][EMB] bf16
              const __bf16* __restrict__ Alo,    // [BATCH][EMB] bf16
              float* __restrict__ out)           // [BATCH][NUM_NODES] f32
{
    const int lane = threadIdx.x & 31;
    const int wave = threadIdx.x >> 5;   // 0..7
    const int half = lane >> 4;          // 0 or 1
    const int lq   = lane & 15;

    const int nb   = blockIdx.x * 16;    // N-tile base (exact: 100000/16 = 6250)
    const int ncol = nb + lq;

    // ---- Load + split B for all 4 K-chunks (register-resident, once per wave).
    // bf16 B 32x16 layout: lane holds column ncol, K = k0 + half*16 + (0..15),
    // i.e. 16 consecutive floats of the embeddings row -> natural row loads.
    v16bf bhi[4], blo[4];
    {
        const float* brow = emb + (size_t)ncol * EMB + half * 16;
        #pragma unroll
        for (int c = 0; c < 4; ++c) {
            float bv[16];
            const float4* p = (const float4*)(brow + c * 32);
            *(float4*)(bv + 0)  = p[0];
            *(float4*)(bv + 4)  = p[1];
            *(float4*)(bv + 8)  = p[2];
            *(float4*)(bv + 12) = p[3];
            #pragma unroll
            for (int i = 0; i < 16; ++i) {
                __bf16 h, l;
                split_bf16(bv[i], h, l);
                bhi[c][i] = h;
                blo[c][i] = l;
            }
        }
    }

    // ---- 8 M-tiles per wave; A arrives pre-split, no conversion VALU here.
    #pragma unroll 1
    for (int t = 0; t < 8; ++t) {
        const int mrow = wave * 128 + t * 16 + lq;
        const __bf16* ahrow = Ahi + (size_t)mrow * EMB;
        const __bf16* alrow = Alo + (size_t)mrow * EMB;

        v8f acc = {};

        #pragma unroll
        for (int c = 0; c < 4; ++c) {
            const int k0 = c * 32;
            // bf16 A 16x32 layout: lane holds row mrow,
            //   elements 0..7  = K = k0 + half*16     + (0..7)
            //   elements 8..15 = K = k0 + half*16 + 8 + (0..7)
            // Each 8-element group is 16 B -> one global_load_b128.
            v8bf ah0 = *(const v8bf*)(ahrow + k0 + half * 16);
            v8bf ah1 = *(const v8bf*)(ahrow + k0 + half * 16 + 8);
            v8bf al0 = *(const v8bf*)(alrow + k0 + half * 16);
            v8bf al1 = *(const v8bf*)(alrow + k0 + half * 16 + 8);
            v16bf ahi = __builtin_shufflevector(ah0, ah1,
                            0, 1, 2, 3, 4, 5, 6, 7, 8, 9, 10, 11, 12, 13, 14, 15);
            v16bf alo = __builtin_shufflevector(al0, al1,
                            0, 1, 2, 3, 4, 5, 6, 7, 8, 9, 10, 11, 12, 13, 14, 15);

            // D = A*B + C with f32 accumulate; split-precision: drop lo*lo.
            acc = __builtin_amdgcn_wmma_f32_16x16x32_bf16(
                      false, ahi, false, bhi[c], (short)0, acc, false, false);
            acc = __builtin_amdgcn_wmma_f32_16x16x32_bf16(
                      false, ahi, false, blo[c], (short)0, acc, false, false);
            acc = __builtin_amdgcn_wmma_f32_16x16x32_bf16(
                      false, alo, false, bhi[c], (short)0, acc, false, false);
        }

        // ---- Store C/D: VGPR r -> row mbase + r, col ncol.
        // Half-wave writes 16 consecutive floats (64 B) per r -> coalesced.
        const int mbase = wave * 128 + t * 16 + half * 8;
        #pragma unroll
        for (int r = 0; r < 8; ++r) {
            out[(size_t)(mbase + r) * NUM_NODES + ncol] = acc[r];
        }
    }
}

// ---------------------------------------------------------------------------
// Launch
// ---------------------------------------------------------------------------
extern "C" void kernel_launch(void* const* d_in, const int* in_sizes, int n_in,
                              void* d_out, int out_size, void* d_ws, size_t ws_size,
                              hipStream_t stream)
{
    const int*   subj = (const int*)d_in[0];    // batch_subj_index [1024]
    const int*   rel  = (const int*)d_in[1];    // rel_index        [1024]
    const float* emb  = (const float*)d_in[2];  // embeddings [100000][128]
    const float* rels = (const float*)d_in[3];  // relations  [18][128]
    float*       out  = (float*)d_out;          // [1024][100000]

    __bf16* Ahi = (__bf16*)d_ws;                // 256 KB
    __bf16* Alo = Ahi + (size_t)BATCH * EMB;    // 256 KB

    distmult_combine_split<<<(BATCH * EMB) / 256, 256, 0, stream>>>(
        subj, rel, emb, rels, Ahi, Alo);
    distmult_gemm<<<NUM_NODES / 16, 256, 0, stream>>>(emb, Ahi, Alo, out);
}
